// EmbeddingBlock_77524159693443
// MI455X (gfx1250) — compile-verified, matrix operand order checked by reading, not run on hardware
//
#include <hip/hip_runtime.h>

typedef __attribute__((ext_vector_type(16))) _Float16 v16h;
typedef __attribute__((ext_vector_type(8)))  _Float16 v8h;
typedef __attribute__((ext_vector_type(8)))  float    v8f;
typedef int v4i_b __attribute__((vector_size(4 * sizeof(int))));

#define N_RAD      6
#define HID        128
#define KTOT       384          // 3*HID
#define KSTEPS     12           // KTOT / 32
#define NTILES     8            // HID / 16
#define VOCAB_N    95
#define EMB_PAD    136          // 128 + 8 halves: 272B row stride breaks bank conflicts
#define EPW        32           // edges per wave = two 16-row M tiles (B fragment reuse x2)
#define WAVES      8
#define EPB        (EPW*WAVES)  // 256 edges per block
#define B_HALVES   (KSTEPS*NTILES*32*16)   // 49152 halves = 96 KB
#define NFRAG      (KSTEPS*NTILES)         // 96 B fragments

#if __has_builtin(__builtin_amdgcn_global_load_async_to_lds_b128)
#define HAVE_ASYNC_LDS 1
#else
#define HAVE_ASYNC_LDS 0
#endif

// A/B 16-bit WMMA fragment K index for vector element e (0..15) given lane half.
__device__ __forceinline__ int kmap(int e, int hf) {
    return 8*hf + e + ((e >= 8) ? 8 : 0);
}

__device__ __forceinline__ float fast_swish(float y) {
#if __has_builtin(__builtin_amdgcn_exp2f) && __has_builtin(__builtin_amdgcn_rcpf)
    // sigmoid(y) = rcp(1 + exp2(-y*log2(e)))  -> v_exp_f32 + v_rcp_f32
    float e = __builtin_amdgcn_exp2f(y * -1.442695041f);
    return y * __builtin_amdgcn_rcpf(1.0f + e);
#else
    return y / (1.0f + __expf(-y));
#endif
}

// ---------------------------------------------------------------------------
// Kernel 0: repack W_lin [128, 384] fp32 (row n = output channel, col k) into
// f16 B-fragments: layout [kstep][ntile][lane][16 halves], B[k][n] = W_lin[n][k].
// ---------------------------------------------------------------------------
__global__ void repack_wlin_kernel(const float* __restrict__ Wlin,
                                   _Float16* __restrict__ Bp) {
    int idx = blockIdx.x * blockDim.x + threadIdx.x;
    if (idx >= B_HALVES) return;
    int e    = idx & 15;
    int lane = (idx >> 4) & 31;
    int nt   = (idx >> 9) & 7;
    int ks   = idx >> 12;           // 0..11
    int hf   = lane >> 4;
    int k    = 32*ks + kmap(e, hf);
    int n    = 16*nt + (lane & 15);
    Bp[idx] = (_Float16)Wlin[n * KTOT + k];
}

// ---------------------------------------------------------------------------
// Fused embedding-block kernel. Each wave owns a 32-edge x 128-col output
// tile (two 16x16x32 M tiles sharing every B fragment -> 192 WMMA / wave),
// with explicit double-buffering of A and B fragment streams.
// ---------------------------------------------------------------------------
__global__ __launch_bounds__(256)
__attribute__((amdgpu_waves_per_eu(4)))
void emb_block_kernel(const int*   __restrict__ x,
                      const float* __restrict__ rbf,
                      const int*   __restrict__ eidx_i,
                      const int*   __restrict__ eidx_j,
                      const float* __restrict__ emb,
                      const float* __restrict__ Wrbf,
                      const float* __restrict__ brbf,
                      const _Float16* __restrict__ Bp,
                      const float* __restrict__ blin,
                      float* __restrict__ out,
                      int nEdges) {
    __shared__ _Float16 sB[B_HALVES];            // 96 KB repacked W_lin fragments
    __shared__ _Float16 sEmb[VOCAB_N * EMB_PAD]; // ~25.8 KB f16 embedding table
    __shared__ float    sWrbf[HID * N_RAD];      // 3 KB
    __shared__ float    sbrbf[HID];
    __shared__ float    sblin[HID];

    const int tid = threadIdx.x;

    // ---- cooperative staging --------------------------------------------
#if HAVE_ASYNC_LDS
    for (int t = tid; t < B_HALVES / 8; t += 256) {
        __builtin_amdgcn_global_load_async_to_lds_b128(
            (__attribute__((address_space(1))) v4i_b*)(Bp + (size_t)t * 8),
            (__attribute__((address_space(3))) v4i_b*)(sB + (size_t)t * 8),
            0, 0);
    }
#else
    {
        const uint4* src = (const uint4*)Bp;     // 16B chunks
        uint4*       dst = (uint4*)sB;
        for (int t = tid; t < B_HALVES / 8; t += 256) dst[t] = src[t];
    }
#endif
    for (int t = tid; t < VOCAB_N * HID; t += 256) {
        int r = t >> 7, c = t & 127;
        sEmb[r * EMB_PAD + c] = (_Float16)emb[t];
    }
    for (int t = tid; t < HID * N_RAD; t += 256) sWrbf[t] = Wrbf[t];
    for (int t = tid; t < HID; t += 256) { sbrbf[t] = brbf[t]; sblin[t] = blin[t]; }
#if HAVE_ASYNC_LDS
#if __has_builtin(__builtin_amdgcn_s_wait_asynccnt)
    __builtin_amdgcn_s_wait_asynccnt(0);
#else
    asm volatile("s_wait_asynccnt 0x0" ::: "memory");
#endif
#endif
    __syncthreads();

    // ---- per-wave 32-edge x 128-col tile --------------------------------
    const int wave = tid >> 5;
    const int lane = tid & 31;
    const int hf   = lane >> 4;       // lane half selects K sub-blocks / M offset
    const int row  = lane & 15;       // edge-in-tile (A build) and N-in-tile (B/D)

    const long eBase = (long)blockIdx.x * EPB + wave * EPW;
    long eC[2];                       // this lane's (clamped) edge in M-tile 0 / 1
    eC[0] = eBase + row;
    eC[1] = eBase + 16 + row;
    eC[0] = (eC[0] < nEdges) ? eC[0] : (long)(nEdges - 1);
    eC[1] = (eC[1] < nEdges) ? eC[1] : (long)(nEdges - 1);

    // gather node types + rbf rows for both tiles
    long embI[2], embJ[2];
    float r6[2][N_RAD];
#pragma unroll
    for (int t = 0; t < 2; ++t) {
        embI[t] = (long)x[eidx_i[eC[t]]] * EMB_PAD;
        embJ[t] = (long)x[eidx_j[eC[t]]] * EMB_PAD;
#pragma unroll
        for (int r = 0; r < N_RAD; ++r) r6[t][r] = rbf[eC[t] * N_RAD + r];
    }

    v8f acc[2][NTILES];
#pragma unroll
    for (int t = 0; t < 2; ++t)
#pragma unroll
        for (int nt = 0; nt < NTILES; ++nt)
            acc[t][nt] = (v8f){0.f,0.f,0.f,0.f,0.f,0.f,0.f,0.f};

    union AFrag { v16h v; v8h h2[2]; _Float16 e[16]; };
    AFrag aF[2][2];                   // [buf][tile] ping-pong A fragments
    v16h  bb[2];                      // ping-pong B fragments

    // B fragment for lane at linear index li (= ks*NTILES + nt)
#define LOADB(li) (*(const v16h*)&sB[(size_t)(li) * 512 + (size_t)lane * 16])

    // A for ks = 0 (emb_i, K=0..31)
    aF[0][0].h2[0] = *(const v8h*)&sEmb[embI[0] + 8*hf];
    aF[0][0].h2[1] = *(const v8h*)&sEmb[embI[0] + 16 + 8*hf];
    aF[0][1].h2[0] = *(const v8h*)&sEmb[embI[1] + 8*hf];
    aF[0][1].h2[1] = *(const v8h*)&sEmb[embI[1] + 16 + 8*hf];
    bb[0] = LOADB(0);

#pragma unroll
    for (int ks = 0; ks < KSTEPS; ++ks) {
        const int cur = ks & 1;
        const int nxt = cur ^ 1;
        // ---- prefetch / precompute A for ks+1 while ks's WMMAs drain ----
        if (ks < 7) {
            const int  ksn = ks + 1;
            const long rb0 = (ksn < 4) ? embI[0] : embJ[0];
            const long rb1 = (ksn < 4) ? embI[1] : embJ[1];
            const int  k0  = 32 * (ksn & 3);
            aF[nxt][0].h2[0] = *(const v8h*)&sEmb[rb0 + k0 + 8*hf];
            aF[nxt][0].h2[1] = *(const v8h*)&sEmb[rb0 + k0 + 16 + 8*hf];
            aF[nxt][1].h2[0] = *(const v8h*)&sEmb[rb1 + k0 + 8*hf];
            aF[nxt][1].h2[1] = *(const v8h*)&sEmb[rb1 + k0 + 16 + 8*hf];
        } else if (ks < 11) {
            const int kr = 32 * (ks + 1 - 8);   // rbf_h channels for ks+1
#pragma unroll
            for (int e2 = 0; e2 < 16; ++e2) {
                const int c = kr + kmap(e2, hf);       // channel 0..127
                const float* w = &sWrbf[c * N_RAD];
                const float bbias = sbrbf[c];
                float s0 = bbias, s1 = bbias;
#pragma unroll
                for (int r = 0; r < N_RAD; ++r) {
                    s0 += r6[0][r] * w[r];
                    s1 += r6[1][r] * w[r];
                }
                aF[nxt][0].e[e2] = (_Float16)fast_swish(s0);
                aF[nxt][1].e[e2] = (_Float16)fast_swish(s1);
            }
        }
        // ---- 8 N-tiles, B double-buffered -------------------------------
#pragma unroll
        for (int nt = 0; nt < NTILES; ++nt) {
            const int li = ks * NTILES + nt;
            const int bc = li & 1;
            bb[bc ^ 1] = LOADB((li < NFRAG - 1) ? li + 1 : NFRAG - 1);
            acc[0][nt] = __builtin_amdgcn_wmma_f32_16x16x32_f16(
                false, aF[cur][0].v, false, bb[bc], (short)0, acc[0][nt], false, false);
            acc[1][nt] = __builtin_amdgcn_wmma_f32_16x16x32_f16(
                false, aF[cur][1].v, false, bb[bc], (short)0, acc[1][nt], false, false);
        }
    }
#undef LOADB

    // ---- epilogue: bias + swish + store ---------------------------------
    // D layout: lane (hf,row), VGPR v -> element (M = v + 8*hf, N = 16*nt + row)
#pragma unroll
    for (int t = 0; t < 2; ++t) {
#pragma unroll
        for (int nt = 0; nt < NTILES; ++nt) {
            const int n = nt * 16 + row;
            const float bl = sblin[n];
#pragma unroll
            for (int v = 0; v < 8; ++v) {
                const int  m  = v + 8 * hf;
                const long eo = eBase + 16 * t + m;
                float y = fast_swish(acc[t][nt][v] + bl);
                if (eo < nEdges) out[eo * (long)HID + n] = y;
            }
        }
    }
}

// ---------------------------------------------------------------------------
extern "C" void kernel_launch(void* const* d_in, const int* in_sizes, int n_in,
                              void* d_out, int out_size, void* d_ws, size_t ws_size,
                              hipStream_t stream) {
    const int*   x    = (const int*)  d_in[0];
    const float* rbf  = (const float*)d_in[1];
    const int*   ei   = (const int*)  d_in[2];
    const int*   ej   = (const int*)  d_in[3];
    const float* emb  = (const float*)d_in[4];
    const float* Wrbf = (const float*)d_in[5];
    const float* brbf = (const float*)d_in[6];
    const float* Wlin = (const float*)d_in[7];
    const float* blin = (const float*)d_in[8];
    float* out = (float*)d_out;
    const int nEdges = in_sizes[2];

    _Float16* Bp = (_Float16*)d_ws;   // 96 KB repacked W_lin fragments

    repack_wlin_kernel<<<(B_HALVES + 255) / 256, 256, 0, stream>>>(Wlin, Bp);

    const int blocks = (nEdges + EPB - 1) / EPB;
    emb_block_kernel<<<blocks, 256, 0, stream>>>(
        x, rbf, ei, ej, emb, Wrbf, brbf, Bp, blin, out, nEdges);
}
